// GreaseLM_45320494907769
// MI455X (gfx1250) — compile-verified, matrix operand order checked by reading, not run on hardware
//
#include <hip/hip_runtime.h>

// ---------------- types ----------------
typedef __bf16 bf16_t;
typedef __attribute__((ext_vector_type(16))) __bf16 v16bf;
typedef __attribute__((ext_vector_type(8)))  float  v8f;
typedef long long ll;

__device__ __forceinline__ float gelu_f(float x) {
    return 0.5f * x * (1.0f + erff(x * 0.70710678118654752f));
}

// monotone float<->uint encoding for atomicMax on floats (handles negatives)
__device__ __forceinline__ unsigned fenc(float f) {
    unsigned u = __float_as_uint(f);
    return (u & 0x80000000u) ? ~u : (u | 0x80000000u);
}
__device__ __forceinline__ float fdec(unsigned u) {
    unsigned b = (u & 0x80000000u) ? (u & 0x7fffffffu) : ~u;
    return __uint_as_float(b);
}

// ---------------- generic batched WMMA GEMM ----------------
// C[z] = A[z] @ B[z] (+bias, epilogue), A: MxK fp32, B: KxN fp32 (or NxK if TRANSB)
// Tiling: block 256x128, 8 waves in 4x2, each wave owns a 64x64 tile
//   -> 16 v_wmma_f32_16x16x32_bf16 per K-step per wave, 16 ds_load_b128 (1.0 ld/wmma)
// AMODE 0: A[row*lda+k]
// AMODE 1: row r -> concat( X[gidx[r]][0:200], Mg[gidx[r]][0:200], Eg[r][0:200] )  (K=600)
// AMODE 2: row r -> concat( X[gidx[r]][0:200], Mg[gidx[r]][0:200] )                (K=400)
// EPI 0: none; 1: +bias; 2: gelu(+bias); 3: *scale + mask[zb*maskld+col]
// NOTE: all K and N used here are multiples of 4, so 4-wide groups never straddle
// a bound or a gather-concat segment boundary -> group-level guards only.
#define BM 256
#define BN 128
#define BK 32
#define LDKP 40   // padded K stride (bf16 elems): 80B rows, 8B-aligned 4-groups
#define ASZ (BM * LDKP)
#define BSZ (BN * LDKP)

template<int AMODE, bool TRANSB, int EPI>
__global__ __launch_bounds__(256) void gemm_wmma(
    int M, int N, int K,
    const float* __restrict__ A, int lda, ll sA1, ll sA2,
    const float* __restrict__ B, int ldb, ll sB1, ll sB2,
    const float* __restrict__ bias,
    float* __restrict__ C, int ldc, ll sC1, ll sC2,
    int HH, float scale, const float* __restrict__ mask, int maskld,
    const float* __restrict__ Xg, const float* __restrict__ Eg,
    const float* __restrict__ Mg, const int* __restrict__ gidx)
{
    __shared__ bf16_t As[2 * ASZ];
    __shared__ bf16_t Bs[2 * BSZ];

    const int t  = threadIdx.x;
    const int m0 = blockIdx.y * BM;
    const int n0 = blockIdx.x * BN;
    const int z  = blockIdx.z;
    const int zb = z / HH, zh = z % HH;

    const float* Ab = (AMODE == 0) ? (A + zb * sA1 + zh * sA2) : nullptr;
    const float* Bb = B + zb * sB1 + zh * sB2;
    float*       Cb = C + zb * sC1 + zh * sC2;

    const int lane = t & 31, w = t >> 5;
    const int wr = w >> 1, wc = w & 1;   // 4x2 wave grid, 64x64 per wave

    // ---- register-staged tile loaders (float4 global loads) ----
    auto loadA = [&](int k0, float4* ra) {
#pragma unroll
        for (int i = 0; i < 8; ++i) {
            int idx4 = i * 256 + t;
            int r = idx4 >> 3, kq = (idx4 & 7) * 4;
            int gr = m0 + r, gk = k0 + kq;
            float4 v = {0.f, 0.f, 0.f, 0.f};
            if (gr < M && gk < K) {
                if (AMODE == 0) {
                    v = *(const float4*)&Ab[(ll)gr * lda + gk];
                } else if (AMODE == 1 && gk >= 400) {
                    v = *(const float4*)&Eg[(ll)gr * 200 + gk - 400];
                } else {
                    int node = gidx[gr];
                    v = (gk < 200) ? *(const float4*)&Xg[(ll)node * 200 + gk]
                                   : *(const float4*)&Mg[(ll)node * 200 + gk - 200];
                }
            }
            ra[i] = v;
        }
    };
    auto storeA = [&](int buf, const float4* ra) {
        bf16_t* dst = &As[buf * ASZ];
#pragma unroll
        for (int i = 0; i < 8; ++i) {
            int idx4 = i * 256 + t;
            int r = idx4 >> 3, kq = (idx4 & 7) * 4;
            union { bf16_t h[4]; uint2 u; } pk;
            pk.h[0] = (bf16_t)ra[i].x; pk.h[1] = (bf16_t)ra[i].y;
            pk.h[2] = (bf16_t)ra[i].z; pk.h[3] = (bf16_t)ra[i].w;
            *(uint2*)&dst[r * LDKP + kq] = pk.u;
        }
    };
    // TRANSB=false: per-thread 4x4 micro-tile: 4 coalesced float4 row loads,
    // register transpose, 4 packed b64 LDS stores into Bs[n][k].
    auto loadB = [&](int k0, float4* rb) {
        if (TRANSB) {
#pragma unroll
            for (int i = 0; i < 4; ++i) {
                int idx4 = i * 256 + t;
                int nn = idx4 >> 3, kq = (idx4 & 7) * 4;
                int gn = n0 + nn, gk = k0 + kq;
                float4 v = {0.f, 0.f, 0.f, 0.f};
                if (gn < N && gk < K) v = *(const float4*)&Bb[(ll)gn * ldb + gk];
                rb[i] = v;
            }
        } else {
            int kq4 = (t >> 5) * 4;      // 0..28
            int nq  = (t & 31) * 4;      // 0..124
#pragma unroll
            for (int j = 0; j < 4; ++j) {
                int gk = k0 + kq4 + j, gn = n0 + nq;
                float4 v = {0.f, 0.f, 0.f, 0.f};
                if (gn < N && gk < K) v = *(const float4*)&Bb[(ll)gk * ldb + gn];
                rb[j] = v;
            }
        }
    };
    auto storeB = [&](int buf, const float4* rb) {
        bf16_t* dst = &Bs[buf * BSZ];
        if (TRANSB) {
#pragma unroll
            for (int i = 0; i < 4; ++i) {
                int idx4 = i * 256 + t;
                int nn = idx4 >> 3, kq = (idx4 & 7) * 4;
                union { bf16_t h[4]; uint2 u; } pk;
                pk.h[0] = (bf16_t)rb[i].x; pk.h[1] = (bf16_t)rb[i].y;
                pk.h[2] = (bf16_t)rb[i].z; pk.h[3] = (bf16_t)rb[i].w;
                *(uint2*)&dst[nn * LDKP + kq] = pk.u;
            }
        } else {
            int kq4 = (t >> 5) * 4;
            int nq  = (t & 31) * 4;
#pragma unroll
            for (int c = 0; c < 4; ++c) {
                union { bf16_t h[4]; uint2 u; } pk;
                pk.h[0] = (bf16_t)((const float*)&rb[0])[c];
                pk.h[1] = (bf16_t)((const float*)&rb[1])[c];
                pk.h[2] = (bf16_t)((const float*)&rb[2])[c];
                pk.h[3] = (bf16_t)((const float*)&rb[3])[c];
                *(uint2*)&dst[(nq + c) * LDKP + kq4] = pk.u;
            }
        }
    };

    v8f acc[4][4];
#pragma unroll
    for (int mi = 0; mi < 4; ++mi)
#pragma unroll
        for (int ni = 0; ni < 4; ++ni)
#pragma unroll
            for (int e = 0; e < 8; ++e) acc[mi][ni][e] = 0.0f;

    float4 ra[8], rb[4];
    loadA(0, ra); loadB(0, rb);
    storeA(0, ra); storeB(0, rb);
    __syncthreads();

    int p = 0;
    for (int k0 = 0; k0 < K; k0 += BK) {
        const int kn = k0 + BK;
        const bool more = (kn < K);
        if (more) { loadA(kn, ra); loadB(kn, rb); }  // pipelined prefetch into regs

        // ---- fragments per ISA 7.12.2: lane<16 -> K {0..7,16..23}, lane>=16 -> {8..15,24..31}
        const int lr = lane & 15;
        const int lo = (lane >> 4) * 8;
        const bf16_t* ab = &As[p * ASZ];
        const bf16_t* bb = &Bs[p * BSZ];
        v16bf aF[4], bF[4];
#pragma unroll
        for (int mi = 0; mi < 4; ++mi) {
            const bf16_t* q = &ab[(wr * 64 + mi * 16 + lr) * LDKP + lo];
#pragma unroll
            for (int j = 0; j < 8; ++j) { aF[mi][j] = q[j]; aF[mi][8 + j] = q[16 + j]; }
        }
#pragma unroll
        for (int ni = 0; ni < 4; ++ni) {
            const bf16_t* q = &bb[(wc * 64 + ni * 16 + lr) * LDKP + lo];
#pragma unroll
            for (int j = 0; j < 8; ++j) { bF[ni][j] = q[j]; bF[ni][8 + j] = q[16 + j]; }
        }
#pragma unroll
        for (int mi = 0; mi < 4; ++mi)
#pragma unroll
            for (int ni = 0; ni < 4; ++ni)
                acc[mi][ni] = __builtin_amdgcn_wmma_f32_16x16x32_bf16(
                    false, aF[mi], false, bF[ni], (short)0, acc[mi][ni], false, false);

        if (more) { storeA(p ^ 1, ra); storeB(p ^ 1, rb); }
        __syncthreads();
        p ^= 1;
    }

    // ---- epilogue: D layout: vgpr i -> row i (lanes 0-15) / 8+i (lanes 16-31), col = lane&15
    const int lr = lane & 15;
    const int rsel = (lane >> 4) * 8;
    const bool fullC = (m0 + BM <= M) && (n0 + BN <= N);
    if (fullC) {
#pragma unroll
        for (int mi = 0; mi < 4; ++mi)
#pragma unroll
            for (int ni = 0; ni < 4; ++ni)
#pragma unroll
                for (int vi = 0; vi < 8; ++vi) {
                    int row = m0 + wr * 64 + mi * 16 + rsel + vi;
                    int col = n0 + wc * 64 + ni * 16 + lr;
                    float v = acc[mi][ni][vi];
                    if (EPI == 1 || EPI == 2) v += bias[col];
                    if (EPI == 2) v = gelu_f(v);
                    if (EPI == 3) v = v * scale + (mask ? mask[(ll)zb * maskld + col] : 0.0f);
                    Cb[(ll)row * ldc + col] = v;
                }
    } else {
#pragma unroll
        for (int mi = 0; mi < 4; ++mi)
#pragma unroll
            for (int ni = 0; ni < 4; ++ni)
#pragma unroll
                for (int vi = 0; vi < 8; ++vi) {
                    int row = m0 + wr * 64 + mi * 16 + rsel + vi;
                    int col = n0 + wc * 64 + ni * 16 + lr;
                    if (row < M && col < N) {
                        float v = acc[mi][ni][vi];
                        if (EPI == 1 || EPI == 2) v += bias[col];
                        if (EPI == 2) v = gelu_f(v);
                        if (EPI == 3) v = v * scale + (mask ? mask[(ll)zb * maskld + col] : 0.0f);
                        Cb[(ll)row * ldc + col] = v;
                    }
                }
    }
}

// ---------------- softmax over rows of 256 (one wave32 per row) ----------------
__global__ __launch_bounds__(256) void softmax256(float* __restrict__ S, int rows) {
    int w = threadIdx.x >> 5, lane = threadIdx.x & 31;
    int row = blockIdx.x * 8 + w;
    if (row >= rows) return;
    float* p = S + (ll)row * 256;
    float v[8], mx = -3.0e38f;
#pragma unroll
    for (int i = 0; i < 8; ++i) { v[i] = p[lane + i * 32]; mx = fmaxf(mx, v[i]); }
#pragma unroll
    for (int o = 16; o > 0; o >>= 1) mx = fmaxf(mx, __shfl_xor(mx, o, 32));
    float s = 0.0f;
#pragma unroll
    for (int i = 0; i < 8; ++i) { v[i] = expf(v[i] - mx); s += v[i]; }
#pragma unroll
    for (int o = 16; o > 0; o >>= 1) s += __shfl_xor(s, o, 32);
    float inv = 1.0f / s;
#pragma unroll
    for (int i = 0; i < 8; ++i) p[lane + i * 32] = v[i] * inv;
}

// ---------------- h = LayerNorm(h + add) * g + b, rows of 768 ----------------
__global__ __launch_bounds__(256) void add_ln(float* __restrict__ h, const float* __restrict__ add,
                                              const float* __restrict__ g, const float* __restrict__ b) {
    ll row = blockIdx.x;
    float* hp = h + row * 768;
    const float* ap = add + row * 768;
    int t = threadIdx.x;
    float x0 = hp[t] + ap[t];
    float x1 = hp[t + 256] + ap[t + 256];
    float x2 = hp[t + 512] + ap[t + 512];
    float s = x0 + x1 + x2, s2 = x0 * x0 + x1 * x1 + x2 * x2;
#pragma unroll
    for (int o = 16; o > 0; o >>= 1) { s += __shfl_xor(s, o, 32); s2 += __shfl_xor(s2, o, 32); }
    __shared__ float rs[8], rq[8];
    int w = t >> 5, lane = t & 31;
    if (lane == 0) { rs[w] = s; rq[w] = s2; }
    __syncthreads();
    float S = 0.0f, S2 = 0.0f;
#pragma unroll
    for (int i = 0; i < 8; ++i) { S += rs[i]; S2 += rq[i]; }
    float mean = S * (1.0f / 768.0f);
    float var = S2 * (1.0f / 768.0f) - mean * mean;
    float inv = rsqrtf(var + 1e-12f);
    hp[t]       = (x0 - mean) * inv * g[t]       + b[t];
    hp[t + 256] = (x1 - mean) * inv * g[t + 256] + b[t + 256];
    hp[t + 512] = (x2 - mean) * inv * g[t + 512] + b[t + 512];
}

// ---------------- GNN helper kernels ----------------
__global__ void build_edges(const int* __restrict__ ei, int* __restrict__ srcA, int* __restrict__ dstA,
                            int E, int EN) {
    int tid = blockIdx.x * blockDim.x + threadIdx.x;
    if (tid >= EN) return;
    if (tid < E) { srcA[tid] = ei[tid]; dstA[tid] = ei[E + tid]; }
    else         { srcA[tid] = tid - E; dstA[tid] = tid - E; }
}

// edge encoder l1 via one-hot row gather + BN(eval) + ReLU
__global__ __launch_bounds__(256) void edge_he(const int* __restrict__ etype,
                                               const int* __restrict__ srcA, const int* __restrict__ dstA,
                                               const int* __restrict__ ntype,
                                               const float* __restrict__ w1, const float* __restrict__ b1,
                                               const float* __restrict__ bng, const float* __restrict__ bnb,
                                               float* __restrict__ out, int E, int EN) {
    int r = blockIdx.x;
    if (r >= EN) return;
    int et = (r < E) ? etype[r] : 38;          // ET=38 -> self-loop flag column
    int ht = ntype[srcA[r]];
    int tt = ntype[dstA[r]];
    int j = threadIdx.x;
    if (j < 200) {
        float v = w1[et * 200 + j] + w1[(39 + ht) * 200 + j] + w1[(43 + tt) * 200 + j] + b1[j];
        v = v * (bng[j] * 0.9999950000374996f) + bnb[j];   // g / sqrt(1 + 1e-5)
        out[(ll)r * 200 + j] = fmaxf(v, 0.0f);
    }
}

__global__ void gat_scores(const float* __restrict__ q, const float* __restrict__ k,
                           float* __restrict__ sc, int EN) {
    int tid = blockIdx.x * blockDim.x + threadIdx.x;
    if (tid >= EN * 4) return;
    int r = tid >> 2, hd = tid & 3;
    const float* qp = q + (ll)r * 200 + hd * 50;
    const float* kp = k + (ll)r * 200 + hd * 50;
    float s = 0.0f;
#pragma unroll 10
    for (int i = 0; i < 50; ++i) s += qp[i] * kp[i];
    sc[tid] = s * 0.14142135623730951f;  // 1/sqrt(50)
}

__global__ void gat_segmax(const float* __restrict__ sc, const int* __restrict__ srcA,
                           unsigned* __restrict__ smax, int EN) {
    int tid = blockIdx.x * blockDim.x + threadIdx.x;
    if (tid >= EN * 4) return;
    int r = tid >> 2, hd = tid & 3;
    atomicMax(&smax[srcA[r] * 4 + hd], fenc(sc[tid]));
}

__global__ void gat_expsum(float* __restrict__ sc, const int* __restrict__ srcA,
                           const unsigned* __restrict__ smax, float* __restrict__ ssum,
                           float* __restrict__ deg, int EN) {
    int tid = blockIdx.x * blockDim.x + threadIdx.x;
    if (tid >= EN * 4) return;
    int r = tid >> 2, hd = tid & 3;
    int s = srcA[r];
    float e = expf(sc[tid] - fdec(smax[s * 4 + hd]));
    sc[tid] = e;  // overwrite with exp
    atomicAdd(&ssum[s * 4 + hd], e);
    if (hd == 0) atomicAdd(&deg[s], 1.0f);
}

__global__ void gat_scatter(const float* __restrict__ ex, const float* __restrict__ msg,
                            const int* __restrict__ srcA, const int* __restrict__ dstA,
                            const float* __restrict__ ssum, const float* __restrict__ deg,
                            float* __restrict__ Xagg, int EN) {
    ll tid = (ll)blockIdx.x * blockDim.x + threadIdx.x;
    if (tid >= (ll)EN * 200) return;
    int r = (int)(tid / 200), j = (int)(tid % 200), hd = j / 50;
    int s = srcA[r];
    float coef = ex[r * 4 + hd] / (ssum[s * 4 + hd] + 1e-16f) * deg[s];
    atomicAdd(&Xagg[(ll)dstA[r] * 200 + j], msg[(ll)r * 200 + j] * coef);
}

__global__ void gelu_copy(const float* __restrict__ in, float* __restrict__ out, int n) {
    int tid = blockIdx.x * blockDim.x + threadIdx.x;
    if (tid < n) out[tid] = gelu_f(in[tid]);
}

// info exchange between text CLS token and graph context node (16 blocks)
__global__ __launch_bounds__(256) void info_exchange(float* __restrict__ h, float* __restrict__ X,
                                                     const float* __restrict__ w1, const float* __restrict__ b1,
                                                     const float* __restrict__ w2, const float* __restrict__ b2) {
    __shared__ float cc[968];
    __shared__ float t1[200];
    int b = blockIdx.x, t = threadIdx.x;
    for (int j = t; j < 968; j += 256)
        cc[j] = (j < 768) ? h[(ll)b * 256 * 768 + j] : X[(ll)b * 200 * 200 + (j - 768)];
    __syncthreads();
    if (t < 200) {
        float s = b1[t];
        for (int i = 0; i < 968; ++i) s += cc[i] * w1[i * 200 + t];
        t1[t] = gelu_f(s);
    }
    __syncthreads();
    for (int j = t; j < 968; j += 256) {
        float s = b2[j];
        for (int i = 0; i < 200; ++i) s += t1[i] * w2[i * 968 + j];
        if (j < 768) h[(ll)b * 256 * 768 + j] = s;
        else         X[(ll)b * 200 * 200 + (j - 768)] = s;
    }
}

// ---------------- host orchestration ----------------
#define LAUNCH_GEMM(AM, TB, EPI, M, N, K, A, lda, sA1, sA2, B, ldb, sB1, sB2, BIAS, C, ldc, sC1, sC2, HH, Z, SC, MK, MLD, XG, EG, MG, GI) \
    gemm_wmma<AM, TB, EPI><<<dim3(((N) + BN - 1) / BN, ((M) + BM - 1) / BM, (Z)), dim3(256), 0, stream>>>( \
        (M), (N), (K), (A), (lda), (ll)(sA1), (ll)(sA2), (B), (ldb), (ll)(sB1), (ll)(sB2), (BIAS), \
        (C), (ldc), (ll)(sC1), (ll)(sC2), (HH), (SC), (MK), (MLD), (XG), (EG), (MG), (GI))

extern "C" void kernel_launch(void* const* d_in, const int* in_sizes, int n_in,
                              void* d_out, int out_size, void* d_ws, size_t ws_size,
                              hipStream_t stream) {
    const int NL = 12, KG = 5, NH = 12, D = 768, S = 256, BS = 16, DFF = 3072;
    const int E = 32000, NN = 3200, EN = E + NN, CD = 200;
    const int MH = BS * S;  // 4096

    const float* hidden = (const float*)d_in[0];
    const float* amask  = (const float*)d_in[1];
    const float* X0     = (const float*)d_in[3];
    const int*   eidx   = (const int*)d_in[4];
    const int*   etype  = (const int*)d_in[5];
    const int*   ntype  = (const int*)d_in[6];
    const float* extra  = (const float*)d_in[7];
    const float* P      = (const float*)d_in[9];

    float* hptr = (float*)d_out;                 // [4096, 768]
    float* Xptr = hptr + (ll)MH * D;             // [3200, 200]

    // ---- params (JAX pytree order: dict keys sorted alphabetically) ----
    size_t cur = 0;
    auto take = [&](size_t n) { const float* p = P + cur; cur += n; return p; };
    struct BertL { const float *fib,*fiw,*fob,*fow,*kb,*kw,*ln1b,*ln1g,*ln2b,*ln2g,*ob,*ow,*qb,*qw,*vb,*vw; };
    BertL bl[12];
    for (int l = 0; l < NL; ++l) {
        bl[l].fib = take(3072); bl[l].fiw = take((size_t)768 * 3072);
        bl[l].fob = take(768);  bl[l].fow = take((size_t)3072 * 768);
        bl[l].kb  = take(768);  bl[l].kw  = take((size_t)768 * 768);
        bl[l].ln1b = take(768); bl[l].ln1g = take(768);
        bl[l].ln2b = take(768); bl[l].ln2g = take(768);
        bl[l].ob  = take(768);  bl[l].ow  = take((size_t)768 * 768);
        bl[l].qb  = take(768);  bl[l].qw  = take((size_t)768 * 768);
        bl[l].vb  = take(768);  bl[l].vw  = take((size_t)768 * 768);
    }
    const float* bnb = take(200); const float* bng = take(200);
    const float* e1b = take(200); const float* e1w = take((size_t)47 * 200);
    const float* e2b = take(200); const float* e2w = take((size_t)200 * 200);
    struct GnnL { const float *keB,*keW,*msB,*msW,*qrB,*qrW; } gl[5];
    for (int g = 0; g < KG; ++g) {
        gl[g].keB = take(200); gl[g].keW = take((size_t)600 * 200);
        gl[g].msB = take(200); gl[g].msW = take((size_t)600 * 200);
        gl[g].qrB = take(200); gl[g].qrW = take((size_t)400 * 200);
    }
    const float* i1b = take(200); const float* i1w = take((size_t)968 * 200);
    const float* i2b = take(968); const float* i2w = take((size_t)200 * 968);

    // ---- workspace layout (BERT and GNN regions overlap; used in disjoint phases) ----
    char* ws = (char*)d_ws;
    const size_t H4 = (size_t)MH * D * 4;              // 12,582,912
    float* qbuf = (float*)(ws + 0);
    float* kbuf = (float*)(ws + H4);
    float* vbuf = (float*)(ws + 2 * H4);
    float* big0 = (float*)(ws + 3 * H4);               // scores [192,256,256] / FF [4096,3072]
    float* ctxb = (float*)(ws + 3 * H4 + 50331648ull);
    // GNN (overlaps BERT scratch)
    float*    heRelu = (float*)(ws + 0);
    float*    eemb   = (float*)(ws + 28160000ull);
    float*    keybuf = (float*)(ws + 56320000ull);
    float*    msgbuf = (float*)(ws + 84480000ull);
    float*    qrybuf = (float*)(ws + 112640000ull);
    float*    scbuf  = (float*)(ws + 140800000ull);    // [EN,4]
    float*    ssum   = (float*)(ws + 141363200ull);    // [NN,4]
    unsigned* smax   = (unsigned*)(ws + 141926400ull); // [NN,4]
    float*    degb   = (float*)(ws + 142489600ull);    // [NN]
    float*    Xagg   = (float*)(ws + 142502400ull);    // [NN,200]
    int*      srcA   = (int*)(ws + 145062400ull);      // persist past layers
    int*      dstA   = (int*)(ws + 145203200ull);

    // ---- init working outputs ----
    hipMemcpyAsync(hptr, hidden, H4, hipMemcpyDeviceToDevice, stream);
    hipMemcpyAsync(Xptr, X0, (size_t)NN * CD * 4, hipMemcpyDeviceToDevice, stream);
    build_edges<<<(EN + 255) / 256, 256, 0, stream>>>(eidx, srcA, dstA, E, EN);

    for (int L = 0; L < NL; ++L) {
        const BertL& p = bl[L];
        // Q/K/V projections: [4096,768] @ [768,768] + b
        LAUNCH_GEMM(0, false, 1, MH, D, D, hptr, D, 0, 0, p.qw, D, 0, 0, p.qb, qbuf, D, 0, 0, 1, 1, 0.f, (const float*)nullptr, 0, (const float*)nullptr, (const float*)nullptr, (const float*)nullptr, (const int*)nullptr);
        LAUNCH_GEMM(0, false, 1, MH, D, D, hptr, D, 0, 0, p.kw, D, 0, 0, p.kb, kbuf, D, 0, 0, 1, 1, 0.f, (const float*)nullptr, 0, (const float*)nullptr, (const float*)nullptr, (const float*)nullptr, (const int*)nullptr);
        LAUNCH_GEMM(0, false, 1, MH, D, D, hptr, D, 0, 0, p.vw, D, 0, 0, p.vb, vbuf, D, 0, 0, 1, 1, 0.f, (const float*)nullptr, 0, (const float*)nullptr, (const float*)nullptr, (const float*)nullptr, (const int*)nullptr);
        // scores[b,h] = Q_bh (256x64) @ K_bh^T * 0.125 + mask[b]   (Z = 192, HH = 12)
        LAUNCH_GEMM(0, true, 3, S, S, 64,
                    qbuf, D, (ll)S * D, 64,
                    kbuf, D, (ll)S * D, 64,
                    (const float*)nullptr,
                    big0, S, (ll)NH * S * S, (ll)S * S,
                    NH, BS * NH, 0.125f, amask, S,
                    (const float*)nullptr, (const float*)nullptr, (const float*)nullptr, (const int*)nullptr);
        softmax256<<<(BS * NH * S) / 8, 256, 0, stream>>>(big0, BS * NH * S);
        // ctx[b,h] = P_bh (256x256) @ V_bh (256x64)
        LAUNCH_GEMM(0, false, 0, S, 64, S,
                    big0, S, (ll)NH * S * S, (ll)S * S,
                    vbuf, D, (ll)S * D, 64,
                    (const float*)nullptr,
                    ctxb, D, (ll)S * D, 64,
                    NH, BS * NH, 0.f, (const float*)nullptr, 0,
                    (const float*)nullptr, (const float*)nullptr, (const float*)nullptr, (const int*)nullptr);
        // attn output projection -> qbuf (reuse), then h = LN(h + qbuf)
        LAUNCH_GEMM(0, false, 1, MH, D, D, ctxb, D, 0, 0, p.ow, D, 0, 0, p.ob, qbuf, D, 0, 0, 1, 1, 0.f, (const float*)nullptr, 0, (const float*)nullptr, (const float*)nullptr, (const float*)nullptr, (const int*)nullptr);
        add_ln<<<MH, 256, 0, stream>>>(hptr, qbuf, p.ln1g, p.ln1b);
        // FFN: big0 = gelu(h @ fiw + fib); kbuf = big0 @ fow + fob; h = LN(h + kbuf)
        LAUNCH_GEMM(0, false, 2, MH, DFF, D, hptr, D, 0, 0, p.fiw, DFF, 0, 0, p.fib, big0, DFF, 0, 0, 1, 1, 0.f, (const float*)nullptr, 0, (const float*)nullptr, (const float*)nullptr, (const float*)nullptr, (const int*)nullptr);
        LAUNCH_GEMM(0, false, 1, MH, D, DFF, big0, DFF, 0, 0, p.fow, D, 0, 0, p.fob, kbuf, D, 0, 0, 1, 1, 0.f, (const float*)nullptr, 0, (const float*)nullptr, (const float*)nullptr, (const float*)nullptr, (const int*)nullptr);
        add_ln<<<MH, 256, 0, stream>>>(hptr, kbuf, p.ln2g, p.ln2b);

        if (L >= NL - KG) {
            const GnnL& gp = gl[L - (NL - KG)];
            // edge embeddings
            edge_he<<<EN, 256, 0, stream>>>(etype, srcA, dstA, ntype, e1w, e1b, bng, bnb, heRelu, E, EN);
            LAUNCH_GEMM(0, false, 1, EN, CD, CD, heRelu, CD, 0, 0, e2w, CD, 0, 0, e2b, eemb, CD, 0, 0, 1, 1, 0.f, (const float*)nullptr, 0, (const float*)nullptr, (const float*)nullptr, (const float*)nullptr, (const int*)nullptr);
            // key (gather dst), msg (gather src): [EN,600]@[600,200]; qry (gather src): [EN,400]@[400,200]
            LAUNCH_GEMM(1, false, 1, EN, CD, 600, (const float*)nullptr, 0, 0, 0, gp.keW, CD, 0, 0, gp.keB, keybuf, CD, 0, 0, 1, 1, 0.f, (const float*)nullptr, 0, Xptr, eemb, extra, dstA);
            LAUNCH_GEMM(1, false, 1, EN, CD, 600, (const float*)nullptr, 0, 0, 0, gp.msW, CD, 0, 0, gp.msB, msgbuf, CD, 0, 0, 1, 1, 0.f, (const float*)nullptr, 0, Xptr, eemb, extra, srcA);
            LAUNCH_GEMM(2, false, 1, EN, CD, 400, (const float*)nullptr, 0, 0, 0, gp.qrW, CD, 0, 0, gp.qrB, qrybuf, CD, 0, 0, 1, 1, 0.f, (const float*)nullptr, 0, Xptr, eemb, extra, srcA);
            // per-edge attention, segment softmax by src, degree rescale, scatter to dst
            hipMemsetAsync(smax, 0, (size_t)NN * 4 * 4, stream);   // fenc codes are all > 0
            hipMemsetAsync(ssum, 0, (size_t)NN * 4 * 4, stream);
            hipMemsetAsync(degb, 0, (size_t)NN * 4, stream);
            hipMemsetAsync(Xagg, 0, (size_t)NN * CD * 4, stream);
            gat_scores<<<(EN * 4 + 255) / 256, 256, 0, stream>>>(qrybuf, keybuf, scbuf, EN);
            gat_segmax<<<(EN * 4 + 255) / 256, 256, 0, stream>>>(scbuf, srcA, smax, EN);
            gat_expsum<<<(EN * 4 + 255) / 256, 256, 0, stream>>>(scbuf, srcA, smax, ssum, degb, EN);
            gat_scatter<<<(int)(((ll)EN * 200 + 255) / 256), 256, 0, stream>>>(scbuf, msgbuf, srcA, dstA, ssum, degb, Xagg, EN);
            gelu_copy<<<(NN * CD + 255) / 256, 256, 0, stream>>>(Xagg, Xptr, NN * CD);
            // info exchange (CLS token <-> context node)
            info_exchange<<<BS, 256, 0, stream>>>(hptr, Xptr, i1w, i1b, i2w, i2b);
        }
    }
}
